// Encoder_31576599561093
// MI455X (gfx1250) — matrix operand order repaired and checked by reading, not compiled, not run on hardware
//
#include <hip/hip_runtime.h>
#include <hip/hip_bf16.h>

typedef __bf16 bf16;
typedef __attribute__((ext_vector_type(16))) __bf16 v16bf;
typedef __attribute__((ext_vector_type(8)))  __bf16 v8bf;
typedef __attribute__((ext_vector_type(8)))  float  v8f;

// ---------------------------------------------------------------------------
// CDNA5 WMMA 16x16x32 bf16 fragment layouts (wave32):
// A (16x32 MxK): lane's halves 0-7  = K run [k0 + (laneHi?8:0) .. +7]
//                lane's halves 8-15 = K run [k0 + 16 + (laneHi?8:0) .. +7]
//                (M = lane & 15)  -> two contiguous 8-element runs.
// B (32x16 KxN): lane's half h = B[k0 + (h<8?0:16) + (laneHi?8:0) + (h&7)][nTile*16 + (lane&15)]
//                -> pre-packed to fragment order: one aligned 32B load.
// C/D (16x16 f32): VGPR r -> M = r + (laneHi?8:0), N = lane & 15.
// Each wave register-blocks 2 M-tiles (32 rows) sharing one B fragment.
// ---------------------------------------------------------------------------

__device__ __forceinline__ v16bf load_b_frag(const bf16* __restrict__ wf,
                                             int kb, int nTiles, int nTile, int lane) {
  return *(const v16bf*)(wf + ((size_t)((size_t)kb * nTiles + nTile) * 32 + lane) * 16);
}

__device__ __forceinline__ v16bf join_runs(v8bf r0, v8bf r1) {
  v16bf a;
#pragma unroll
  for (int h = 0; h < 8; ++h) { a[h] = r0[h]; a[h + 8] = r1[h]; }
  return a;
}

// ---------------------------------------------------------------------------
// Implicit-GEMM convolution, templated geometry. One wave = 32x16 output tile
// (2 M-tiles x 1 N-tile), block = 128 threads (4 waves -> 4 consecutive
// N tiles). bf16 in/out, f32 accumulate. VEC path requires CIN % 32 == 0
// (a 32-wide K block then never straddles a (kh,kw) cell).
// ---------------------------------------------------------------------------
template <int CIN, int KWT, int PAD, bool VEC>
__global__ __launch_bounds__(128)
void conv_wmma_t(const bf16* __restrict__ in, const bf16* __restrict__ wf,
                 const float* __restrict__ bias, bf16* __restrict__ out,
                 int Hi, int Wi, int Ho, int Wo, int Cout, int K, int Kp,
                 const float* __restrict__ bn_g, const float* __restrict__ bn_b,
                 const float* __restrict__ bn_m, const float* __restrict__ bn_v) {
  const int lane   = threadIdx.x & 31;
  const int wave   = threadIdx.x >> 5;
  const int mBase  = blockIdx.x << 5;            // 32 rows per wave
  const int nTiles = Cout >> 4;
  const int nTile  = blockIdx.y * 4 + wave;
  const int n      = (nTile << 4) + (lane & 15);
  const int laneHiA = (lane & 16) ? 8 : 0;

  // decompose the two A rows this lane services
  int woA[2], hoA[2];
  const bf16* inB[2];
#pragma unroll
  for (int s = 0; s < 2; ++s) {
    int mA = mBase + (s << 4) + (lane & 15);
    woA[s] = mA % Wo;
    int tA = mA / Wo;
    hoA[s] = tA % Ho;
    inB[s] = in + (size_t)(tA / Ho) * Hi * Wi * CIN;
  }

  v8f acc0 = {}, acc1 = {};
  for (int k0 = 0; k0 < Kp; k0 += 32) {
    const int kb = k0 >> 5;
    if (k0 + 32 < Kp)  // prefetch next fragment-packed weight block (gfx1250)
      __builtin_prefetch(wf + ((size_t)((size_t)(kb + 1) * nTiles + nTile) * 32 + lane) * 16, 0, 1);

    v16bf a[2];
    if (VEC) {
      const int cell = k0 / CIN;           // entire 32-block inside one (kh,kw)
      const int ci0  = k0 - cell * CIN;
      const int kw   = cell % KWT;
      const int kh   = cell / KWT;
#pragma unroll
      for (int s = 0; s < 2; ++s) {
        const int yi = hoA[s] + kh - PAD;
        const int xi = woA[s] + kw - PAD;
        v8bf r0 = {}, r1 = {};
        if ((unsigned)yi < (unsigned)Hi && (unsigned)xi < (unsigned)Wi) {
          const bf16* p = inB[s] + ((size_t)yi * Wi + xi) * CIN + ci0 + laneHiA;
          r0 = *(const v8bf*)p;
          r1 = *(const v8bf*)(p + 16);
        }
        a[s] = join_runs(r0, r1);
      }
    } else {
#pragma unroll
      for (int s = 0; s < 2; ++s) {
#pragma unroll
        for (int h = 0; h < 16; ++h) {
          int ka = k0 + ((h < 8) ? 0 : 16) + laneHiA + (h & 7);
          float av = 0.f;
          if (ka < K) {
            int ci = ka % CIN;
            int r  = ka / CIN;
            int kw = r % KWT;
            int kh = r / KWT;
            int yi = hoA[s] + kh - PAD;
            int xi = woA[s] + kw - PAD;
            if ((unsigned)yi < (unsigned)Hi && (unsigned)xi < (unsigned)Wi)
              av = (float)inB[s][((size_t)yi * Wi + xi) * CIN + ci];
          }
          a[s][h] = (bf16)av;
        }
      }
    }
    v16bf b = load_b_frag(wf, kb, nTiles, nTile, lane);
    acc0 = __builtin_amdgcn_wmma_f32_16x16x32_bf16(false, a[0], false, b,
                                                   (short)0, acc0, false, false);
    acc1 = __builtin_amdgcn_wmma_f32_16x16x32_bf16(false, a[1], false, b,
                                                   (short)0, acc1, false, false);
  }

  const float bb = bias[n];
  float sc = 1.f, sh = 0.f;
  if (bn_g) {
    sc = bn_g[n] * rsqrtf(bn_v[n] + 1e-3f);
    sh = bn_b[n] - bn_m[n] * sc;
  }
#pragma unroll
  for (int r = 0; r < 8; ++r) {
    int m0 = mBase + r + ((lane & 16) ? 8 : 0);
    float v0 = acc0[r] + bb;
    v0 = v0 > 0.f ? v0 : 0.f;
    v0 = v0 * sc + sh;
    out[(size_t)m0 * Cout + n] = (bf16)v0;
    float v1 = acc1[r] + bb;
    v1 = v1 > 0.f ? v1 : 0.f;
    v1 = v1 * sc + sh;
    out[(size_t)(m0 + 16) * Cout + n] = (bf16)v1;
  }
}

// ---------------------------------------------------------------------------
// 2x2 maxpool; ZeroPadding2D((0,1)) + VALID pool == OOB width reads are 0.0.
// ---------------------------------------------------------------------------
__global__ void maxpool_kernel(const bf16* __restrict__ in, bf16* __restrict__ out,
                               int Hi, int Wi, int C, int Ho, int Wo,
                               int sH, int sW, int padL, size_t total) {
  size_t i = (size_t)blockIdx.x * blockDim.x + threadIdx.x;
  if (i >= total) return;
  int c  = (int)(i % C);
  size_t r = i / C;
  int wo = (int)(r % Wo); r /= Wo;
  int ho = (int)(r % Ho);
  int b  = (int)(r / Ho);
  float mv = -3.4e38f;
#pragma unroll
  for (int dy = 0; dy < 2; ++dy)
#pragma unroll
    for (int dx = 0; dx < 2; ++dx) {
      int yi = ho * sH + dy;
      int xi = wo * sW + dx - padL;
      float v = 0.f;
      if ((unsigned)xi < (unsigned)Wi)
        v = (float)in[(((size_t)b * Hi + yi) * Wi + xi) * C + c];
      mv = fmaxf(mv, v);
    }
  out[i] = (bf16)mv;
}

// ---------------------------------------------------------------------------
// GRU step kernels, M = 32 batch rows handled by ONE wave per N-tile group
// (2 M-tiles register-blocked, shared B fragment -> weights streamed once).
// A = [x_t | h] with vectorized 8-element runs (din % 32 == 0 so runs never
// straddle the x/h split).
// ---------------------------------------------------------------------------
__device__ __forceinline__ const bf16* gru_a_ptr(const bf16* __restrict__ xt,
                                                 int xRowStride, int din,
                                                 const bf16* __restrict__ hpart,
                                                 int bA, int k) {
  return (k < din) ? xt + (size_t)bA * xRowStride + k
                   : hpart + (size_t)bA * 256 + (k - din);
}

__global__ __launch_bounds__(128)
void gru_gate_kernel(const bf16* __restrict__ xt, int xRowStride, int din,
                     const bf16* __restrict__ gkf, const float* __restrict__ gb,
                     const float* __restrict__ hf, const bf16* __restrict__ hb,
                     float* __restrict__ u, bf16* __restrict__ rh) {
  const int U = 256, nTiles = 32;
  const int lane  = threadIdx.x & 31;
  const int wave  = threadIdx.x >> 5;
  const int nTile = blockIdx.x * 4 + wave;
  const int n     = (nTile << 4) + (lane & 15);
  const int K     = din + U;
  const int bA0   = lane & 15;
  const int laneHiA = (lane & 16) ? 8 : 0;

  v8f acc0 = {}, acc1 = {};
  for (int k0 = 0; k0 < K; k0 += 32) {
    int b0 = k0 + laneHiA;
    v8bf r0 = *(const v8bf*)gru_a_ptr(xt, xRowStride, din, hb, bA0, b0);
    v8bf r1 = *(const v8bf*)gru_a_ptr(xt, xRowStride, din, hb, bA0, b0 + 16);
    v8bf r2 = *(const v8bf*)gru_a_ptr(xt, xRowStride, din, hb, bA0 + 16, b0);
    v8bf r3 = *(const v8bf*)gru_a_ptr(xt, xRowStride, din, hb, bA0 + 16, b0 + 16);
    v16bf a0 = join_runs(r0, r1);
    v16bf a1 = join_runs(r2, r3);
    v16bf b = load_b_frag(gkf, k0 >> 5, nTiles, nTile, lane);
    acc0 = __builtin_amdgcn_wmma_f32_16x16x32_bf16(false, a0, false, b,
                                                   (short)0, acc0, false, false);
    acc1 = __builtin_amdgcn_wmma_f32_16x16x32_bf16(false, a1, false, b,
                                                   (short)0, acc1, false, false);
  }
  const float bb = gb[n];
#pragma unroll
  for (int r = 0; r < 8; ++r) {
    int m0 = r + ((lane & 16) ? 8 : 0);
#pragma unroll
    for (int s = 0; s < 2; ++s) {
      int m = m0 + (s << 4);
      float acc = s ? acc1[r] : acc0[r];
      float sg = 1.f / (1.f + expf(-(acc + bb)));
      if (n < U) rh[m * U + n] = (bf16)(sg * hf[m * U + n]);
      else       u[m * U + (n - U)] = sg;
    }
  }
}

__global__ __launch_bounds__(128)
void gru_cand_kernel(const bf16* __restrict__ xt, int xRowStride, int din,
                     const bf16* __restrict__ ckf, const float* __restrict__ cb,
                     const float* __restrict__ u, float* __restrict__ hf,
                     bf16* __restrict__ hb, const bf16* __restrict__ rh,
                     bf16* __restrict__ yout, int yRowStride,
                     float* __restrict__ outf, int outRowStride) {
  const int U = 256, nTiles = 16;
  const int lane  = threadIdx.x & 31;
  const int wave  = threadIdx.x >> 5;
  const int nTile = blockIdx.x * 4 + wave;
  const int n     = (nTile << 4) + (lane & 15);
  const int K     = din + U;
  const int bA0   = lane & 15;
  const int laneHiA = (lane & 16) ? 8 : 0;

  v8f acc0 = {}, acc1 = {};
  for (int k0 = 0; k0 < K; k0 += 32) {
    int b0 = k0 + laneHiA;
    v8bf r0 = *(const v8bf*)gru_a_ptr(xt, xRowStride, din, rh, bA0, b0);
    v8bf r1 = *(const v8bf*)gru_a_ptr(xt, xRowStride, din, rh, bA0, b0 + 16);
    v8bf r2 = *(const v8bf*)gru_a_ptr(xt, xRowStride, din, rh, bA0 + 16, b0);
    v8bf r3 = *(const v8bf*)gru_a_ptr(xt, xRowStride, din, rh, bA0 + 16, b0 + 16);
    v16bf a0 = join_runs(r0, r1);
    v16bf a1 = join_runs(r2, r3);
    v16bf b = load_b_frag(ckf, k0 >> 5, nTiles, nTile, lane);
    acc0 = __builtin_amdgcn_wmma_f32_16x16x32_bf16(false, a0, false, b,
                                                   (short)0, acc0, false, false);
    acc1 = __builtin_amdgcn_wmma_f32_16x16x32_bf16(false, a1, false, b,
                                                   (short)0, acc1, false, false);
  }
  const float bb = cb[n];
#pragma unroll
  for (int r = 0; r < 8; ++r) {
    int m0 = r + ((lane & 16) ? 8 : 0);
#pragma unroll
    for (int s = 0; s < 2; ++s) {
      int m = m0 + (s << 4);
      float acc = s ? acc1[r] : acc0[r];
      float c  = tanhf(acc + bb);
      float uu = u[m * U + n];
      float hn = uu * hf[m * U + n] + (1.f - uu) * c;
      hf[m * U + n] = hn;
      hb[m * U + n] = (bf16)hn;
      if (yout) yout[(size_t)m * yRowStride + n] = (bf16)hn;
      if (outf) outf[(size_t)m * outRowStride + n] = hn;
    }
  }
}

// ---------------------------------------------------------------------------
// Elementwise helpers
// ---------------------------------------------------------------------------
__global__ void cast_f32_bf16_kernel(const float* __restrict__ s, bf16* __restrict__ d, size_t n) {
  size_t i = (size_t)blockIdx.x * blockDim.x + threadIdx.x;
  if (i < n) d[i] = (bf16)s[i];
}
// Pack f32 [K][N] row-major weights into bf16 WMMA B-fragment order
// [kb][nTile][lane][16], zero-padding K up to Kp (Kp % 32 == 0).
__global__ void pack_wfrag_kernel(const float* __restrict__ s, bf16* __restrict__ d,
                                  int K, int Kp, int N) {
  size_t i = (size_t)blockIdx.x * blockDim.x + threadIdx.x;
  size_t total = (size_t)Kp * N;  // == (Kp/32)*(N/16)*512
  if (i >= total) return;
  int h    = (int)(i & 15);
  int lane = (int)((i >> 4) & 31);
  size_t r = i >> 9;
  int nT   = (int)(r % (N >> 4));
  int kb   = (int)(r / (N >> 4));
  int k = kb * 32 + ((h < 8) ? 0 : 16) + ((lane & 16) ? 8 : 0) + (h & 7);
  int n = (nT << 4) + (lane & 15);
  d[i] = (bf16)((k < K) ? s[(size_t)k * N + n] : 0.f);
}
__global__ void zero_u32_kernel(unsigned* __restrict__ p, size_t n) {
  size_t i = (size_t)blockIdx.x * blockDim.x + threadIdx.x;
  if (i < n) p[i] = 0u;
}
__global__ void copy_f32_kernel(const float* __restrict__ s, float* __restrict__ d, size_t n) {
  size_t i = (size_t)blockIdx.x * blockDim.x + threadIdx.x;
  if (i < n) d[i] = s[i];
}
__global__ void seqlen_kernel(const int* __restrict__ widths, int* __restrict__ sl, int n) {
  int i = blockIdx.x * blockDim.x + threadIdx.x;
  if (i < n) sl[i] = widths[i] / 4 + 1;
}

// ---------------------------------------------------------------------------
// Host orchestration
// ---------------------------------------------------------------------------
static inline size_t alignup(size_t x) { return (x + 255) & ~(size_t)255; }

extern "C" void kernel_launch(void* const* d_in, const int* in_sizes, int n_in,
                              void* d_out, int out_size, void* d_ws, size_t ws_size,
                              hipStream_t stream) {
  (void)in_sizes; (void)n_in; (void)out_size; (void)ws_size;
  const int Bn = 32, T = 129, U = 256;

  const float* inputs = (const float*)d_in[0];
  const int*   widths = (const int*)d_in[1];
  // d_in[2] = training (unused)
  const float* cw[7] = {(const float*)d_in[3], (const float*)d_in[5], (const float*)d_in[7],
                        (const float*)d_in[9], (const float*)d_in[11], (const float*)d_in[13],
                        (const float*)d_in[15]};
  const float* cbias[7] = {(const float*)d_in[4], (const float*)d_in[6], (const float*)d_in[8],
                           (const float*)d_in[10], (const float*)d_in[12], (const float*)d_in[14],
                           (const float*)d_in[16]};
  const float* bn5[4] = {(const float*)d_in[17], (const float*)d_in[18],
                         (const float*)d_in[19], (const float*)d_in[20]};
  const float* bn6[4] = {(const float*)d_in[21], (const float*)d_in[22],
                         (const float*)d_in[23], (const float*)d_in[24]};
  // GRU params: fw0(25-28) bw0(29-32) fw1(33-36) bw1(37-40), each gk,gb,ck,cb
  const float* ggk[4] = {(const float*)d_in[25], (const float*)d_in[29],
                         (const float*)d_in[33], (const float*)d_in[37]};
  const float* ggb[4] = {(const float*)d_in[26], (const float*)d_in[30],
                         (const float*)d_in[34], (const float*)d_in[38]};
  const float* gck[4] = {(const float*)d_in[27], (const float*)d_in[31],
                         (const float*)d_in[35], (const float*)d_in[39]};
  const float* gcb[4] = {(const float*)d_in[28], (const float*)d_in[32],
                         (const float*)d_in[36], (const float*)d_in[40]};

  // ---- conv layer table: Hi,Wi,Cin,Ho,Wo,Cout,KH,KW,pad ----
  const int CH[7][9] = {
      {32, 512,   3, 32, 512,  64, 3, 3, 1},
      {16, 256,  64, 16, 256, 128, 3, 3, 1},
      { 8, 128, 128,  8, 128, 256, 3, 3, 1},
      { 8, 128, 256,  8, 128, 256, 3, 3, 1},
      { 4, 129, 256,  4, 129, 512, 3, 3, 1},
      { 4, 129, 512,  4, 129, 512, 3, 3, 1},
      { 2, 130, 512,  1, 129, 512, 2, 2, 0}};

  // ---- workspace bump allocator ----
  char* ws = (char*)d_ws;
  size_t off = 0;
  auto take = [&](size_t bytes) { char* p = ws + off; off += alignup(bytes); return p; };

  bf16* wpk[7];
  int Karr[7], Kparr[7];
  for (int i = 0; i < 7; ++i) {
    int K = CH[i][6] * CH[i][7] * CH[i][2];
    int Kp = (K + 31) & ~31;
    Karr[i] = K; Kparr[i] = Kp;
    wpk[i] = (bf16*)take((size_t)Kp * CH[i][5] * sizeof(bf16));
  }
  const int gdin[4] = {512, 512, U, U};
  bf16 *gkp[4], *ckp[4];
  for (int g = 0; g < 4; ++g) {
    gkp[g] = (bf16*)take((size_t)(gdin[g] + U) * 2 * U * sizeof(bf16));
    ckp[g] = (bf16*)take((size_t)(gdin[g] + U) * U * sizeof(bf16));
  }
  bf16* x0  = (bf16*)take((size_t)Bn * 32 * 512 * 3 * sizeof(bf16));
  bf16* R0  = (bf16*)take((size_t)Bn * 32 * 512 * 64 * sizeof(bf16));   // 67 MB ping
  bf16* R1  = (bf16*)take((size_t)16512 * 512 * sizeof(bf16));          // 17 MB pong
  bf16* seq = (bf16*)take((size_t)Bn * T * 512 * sizeof(bf16));
  bf16* y0f = (bf16*)take((size_t)Bn * T * U * sizeof(bf16));
  bf16* y0b = (bf16*)take((size_t)Bn * T * U * sizeof(bf16));
  float* hf = (float*)take((size_t)Bn * U * sizeof(float));
  bf16*  hb = (bf16*)take((size_t)Bn * U * sizeof(bf16));
  float* uB = (float*)take((size_t)Bn * U * sizeof(float));
  bf16*  rh = (bf16*)take((size_t)Bn * U * sizeof(bf16));

  auto ew = [&](size_t n) { return dim3((unsigned)((n + 255) / 256)); };

  // ---- pack weights into WMMA fragment order ----
  for (int i = 0; i < 7; ++i) {
    size_t n = (size_t)Kparr[i] * CH[i][5];
    pack_wfrag_kernel<<<ew(n), 256, 0, stream>>>(cw[i], wpk[i], Karr[i], Kparr[i], CH[i][5]);
  }
  for (int g = 0; g < 4; ++g) {
    int K = gdin[g] + U;
    pack_wfrag_kernel<<<ew((size_t)K * 2 * U), 256, 0, stream>>>(ggk[g], gkp[g], K, K, 2 * U);
    pack_wfrag_kernel<<<ew((size_t)K * U), 256, 0, stream>>>(gck[g], ckp[g], K, K, U);
  }

  // ---- input f32 -> bf16 ----
  cast_f32_bf16_kernel<<<ew((size_t)Bn * 32 * 512 * 3), 256, 0, stream>>>(
      inputs, x0, (size_t)Bn * 32 * 512 * 3);

  auto cgrid = [&](int i) {
    int M = Bn * CH[i][3] * CH[i][4];
    return dim3((unsigned)(M / 32), (unsigned)(CH[i][5] / 64));
  };
#define CONV_CALL(i, CIN, KW, PAD, VEC, src, dst, g, b, m, v)                              \
  conv_wmma_t<CIN, KW, PAD, VEC><<<cgrid(i), 128, 0, stream>>>(                            \
      src, wpk[i], cbias[i], dst, CH[i][0], CH[i][1], CH[i][3], CH[i][4], CH[i][5],        \
      Karr[i], Kparr[i], g, b, m, v)

  auto pool = [&](const bf16* src, bf16* dst, int Hi, int Wi, int C,
                  int Ho, int Wo, int sH, int sW, int padL) {
    size_t n = (size_t)Bn * Ho * Wo * C;
    maxpool_kernel<<<ew(n), 256, 0, stream>>>(src, dst, Hi, Wi, C, Ho, Wo, sH, sW, padL, n);
  };

  // ---- CNN pipeline (ping-pong R0/R1) ----
  CONV_CALL(0,   3, 3, 1, false, x0, R0, nullptr, nullptr, nullptr, nullptr);
  pool(R0, R1, 32, 512, 64, 16, 256, 2, 2, 0);
  CONV_CALL(1,  64, 3, 1, true,  R1, R0, nullptr, nullptr, nullptr, nullptr);
  pool(R0, R1, 16, 256, 128, 8, 128, 2, 2, 0);
  CONV_CALL(2, 128, 3, 1, true,  R1, R0, nullptr, nullptr, nullptr, nullptr);
  CONV_CALL(3, 256, 3, 1, true,  R0, R1, nullptr, nullptr, nullptr, nullptr);
  pool(R1, R0, 8, 128, 256, 4, 129, 2, 1, 1);
  CONV_CALL(4, 256, 3, 1, true,  R0, R1, bn5[0], bn5[1], bn5[2], bn5[3]);
  CONV_CALL(5, 512, 3, 1, true,  R1, R0, bn6[0], bn6[1], bn6[2], bn6[3]);
  pool(R0, R1, 4, 129, 512, 2, 130, 2, 1, 1);
  CONV_CALL(6, 512, 2, 0, true,  R1, seq, nullptr, nullptr, nullptr, nullptr);
#undef CONV_CALL

  // ---- GRU stacks ----
  float* outF  = (float*)d_out;                 // [32,129,512]
  float* sFw0  = outF + (size_t)Bn * T * 512;   // [32,256]
  float* sFw1  = sFw0 + (size_t)Bn * U;         // [32,256]
  int*   slOut = (int*)(sFw1 + (size_t)Bn * U); // [32] int32

  auto zero_h = [&]() {
    zero_u32_kernel<<<ew(Bn * U), 256, 0, stream>>>((unsigned*)hf, (size_t)Bn * U);
    zero_u32_kernel<<<ew(Bn * U / 2), 256, 0, stream>>>((unsigned*)hb, (size_t)Bn * U / 2);
  };
  auto gru_step = [&](int g, const bf16* xbase, int xRowStride, int xStep, int t,
                      bf16* ybase, float* obase, int oOff) {
    const bf16* xt = xbase + (size_t)t * xStep;
    gru_gate_kernel<<<dim3(8), 128, 0, stream>>>(xt, xRowStride, gdin[g], gkp[g], ggb[g],
                                                 hf, hb, uB, rh);
    gru_cand_kernel<<<dim3(4), 128, 0, stream>>>(
        xt, xRowStride, gdin[g], ckp[g], gcb[g], uB, hf, hb, rh,
        ybase ? ybase + (size_t)t * U : nullptr, T * U,
        obase ? obase + (size_t)t * 512 + oOff : nullptr, T * 512);
  };

  // fw0: seq -> y0f, keep last state
  zero_h();
  for (int t = 0; t < T; ++t) gru_step(0, seq, T * 512, 512, t, y0f, nullptr, 0);
  copy_f32_kernel<<<ew(Bn * U), 256, 0, stream>>>(hf, sFw0, (size_t)Bn * U);
  // fw1: y0f -> out[:, :, 0:256], keep last state
  zero_h();
  for (int t = 0; t < T; ++t) gru_step(2, y0f, T * U, U, t, nullptr, outF, 0);
  copy_f32_kernel<<<ew(Bn * U), 256, 0, stream>>>(hf, sFw1, (size_t)Bn * U);
  // bw0: seq (reverse) -> y0b
  zero_h();
  for (int t = T - 1; t >= 0; --t) gru_step(1, seq, T * 512, 512, t, y0b, nullptr, 0);
  // bw1: y0b (reverse) -> out[:, :, 256:512]
  zero_h();
  for (int t = T - 1; t >= 0; --t) gru_step(3, y0b, T * U, U, t, nullptr, outF, 256);

  // seq_len = widths // 4 + 1
  seqlen_kernel<<<1, 32, 0, stream>>>(widths, slOut, Bn);
}
